// LabelPropagation_23063974380174
// MI455X (gfx1250) — compile-verified, matrix operand order checked by reading, not run on hardware
//
#include <hip/hip_runtime.h>
#include <hip/hip_bf16.h>
#include <math.h>

// ---------------------------------------------------------------------------
// Label propagation on MI455X (gfx1250, wave32).
//   GEMM1: G = e e^T   via V_WMMA_F32_16X16X32_F16, fused W=exp(...) epilogue
//   top-20/row selection in LDS, symmetric mask, degree-normalize -> S (f16)
//   Chebyshev semi-iteration for (I - 0.99 S) F = y (spectrum [0.01,1.99]),
//   160 iters, one WMMA GEMM (S @ X) per iter; S (33.5MB f16) is L2-resident.
// ---------------------------------------------------------------------------

typedef __attribute__((ext_vector_type(16))) _Float16 v16h;
typedef __attribute__((ext_vector_type(8)))  _Float16 v8h;
typedef __attribute__((ext_vector_type(8)))  float    v8f;

#define NTOT   4096
#define DFEAT  1600
#define NCLS   64
#define NSUPR  1024
#define NQRYR  3072
#define TOPK_K 20
#define ALPHA_ 0.99f
#define EPSF   2.2204460492503131e-16f
#define CHEB_ITERS 160

#define WMMA_F16(acc, a, b) \
  (acc) = __builtin_amdgcn_wmma_f32_16x16x32_f16(false, (a), false, (b), (short)0, (acc), false, false)

// ---------------- wave32 reductions (deterministic) ------------------------
__device__ inline float wave_sum(float v) {
  for (int o = 16; o > 0; o >>= 1) v += __shfl_xor(v, o, 32);
  return v;
}
__device__ inline void wave_argmax(float& v, int& i) {
  for (int o = 16; o > 0; o >>= 1) {
    float ov = __shfl_xor(v, o, 32);
    int   oi = __shfl_xor(i, o, 32);
    if (ov > v || (ov == v && oi < i)) { v = ov; i = oi; }
  }
}

// ---------------- WMMA fragment loader -------------------------------------
// 16x32 f16 fragment, K contiguous in memory (ld in halfs).
// Lane L: row = row0 + (L&15); K chunks {0-7,16-23} for L<16, {8-15,24-31} else.
union frag_u { v16h v; v8h h[2]; };
__device__ inline v16h load_frag(const _Float16* base, int ld, int row0, int k0, int lane) {
  const _Float16* p = base + (size_t)(row0 + (lane & 15)) * ld + (k0 + ((lane >> 4) << 3));
  frag_u u;
  u.h[0] = *(const v8h*)(p);
  u.h[1] = *(const v8h*)(p + 16);
  return u.v;
}

// ---------------- Stage 1: normalize, f16 copy, row sq ----------------------
__global__ __launch_bounds__(32)
void k_norm(const float* __restrict__ emb, const float* __restrict__ sigma,
            _Float16* __restrict__ E16, float* __restrict__ sq) {
  int r = blockIdx.x, lane = threadIdx.x;
  float inv = 1.0f / (sigma[r] + EPSF);
  float s = 0.0f;
  for (int c = lane; c < DFEAT; c += 32) {
    float e = emb[(size_t)r * DFEAT + c] * inv;
    E16[(size_t)r * DFEAT + c] = (_Float16)e;
    s += e * e;
  }
  s = wave_sum(s);
  if (lane == 0) sq[r] = s;
}

// ---------------- Stage 2: Gram GEMM + fused W epilogue ---------------------
// Wave computes a 32x32 output tile (2x2 WMMA tiles). Block = 4 waves = 64x64.
__global__ __launch_bounds__(128)
void k_gram_w(const _Float16* __restrict__ E, const float* __restrict__ sq,
              float* __restrict__ W) {
  int lane = threadIdx.x & 31, wave = threadIdx.x >> 5;
  int m0 = blockIdx.y * 64 + (wave >> 1) * 32;
  int n0 = blockIdx.x * 64 + (wave & 1) * 32;
  v8f acc[2][2] = {};
  for (int k0 = 0; k0 < DFEAT; k0 += 32) {
    v16h a0 = load_frag(E, DFEAT, m0,      k0, lane);
    v16h a1 = load_frag(E, DFEAT, m0 + 16, k0, lane);
    v16h b0 = load_frag(E, DFEAT, n0,      k0, lane);
    v16h b1 = load_frag(E, DFEAT, n0 + 16, k0, lane);
    WMMA_F16(acc[0][0], a0, b0);
    WMMA_F16(acc[0][1], a0, b1);
    WMMA_F16(acc[1][0], a1, b0);
    WMMA_F16(acc[1][1], a1, b1);
  }
  const float inv2d = 0.5f / (float)DFEAT;   // exp(-((sqi+sqj-2g)/d)/2)
  for (int tm = 0; tm < 2; ++tm) {
    int rowb = m0 + tm * 16 + ((lane >> 4) << 3);
    for (int tn = 0; tn < 2; ++tn) {
      int col = n0 + tn * 16 + (lane & 15);
      float sqc = sq[col];
      for (int r = 0; r < 8; ++r) {
        int row = rowb + r;
        float g = acc[tm][tn][r];
        W[(size_t)row * NTOT + col] = __expf(-(sq[row] + sqc - 2.0f * g) * inv2d);
      }
    }
  }
}

// ---------------- Stage 3: exact top-20 threshold per row (LDS) -------------
__global__ __launch_bounds__(32)
void k_topk(const float* __restrict__ W, float* __restrict__ th) {
  __shared__ float row[NTOT];           // 16 KB of the 320 KB WGP LDS
  int r = blockIdx.x, lane = threadIdx.x;
  for (int c = lane; c < NTOT; c += 32) row[c] = W[(size_t)r * NTOT + c];
  __syncthreads();
  float last = 0.0f;
  for (int t = 0; t < TOPK_K; ++t) {
    float v = -1.0f; int idx = 0;       // W in (0,1], -1 is a safe sentinel
    for (int c = lane; c < NTOT; c += 32) {
      float x = row[c];
      if (x > v) { v = x; idx = c; }
    }
    wave_argmax(v, idx);
    last = v;
    __syncthreads();
    if (lane == 0) row[idx] = -1.0f;
    __syncthreads();
  }
  if (lane == 0) th[r] = last;
}

// ---------------- Stage 4: symmetric mask + degree --------------------------
__global__ __launch_bounds__(32)
void k_mask_deg(float* __restrict__ W, const float* __restrict__ th,
                float* __restrict__ deg) {
  int r = blockIdx.x, lane = threadIdx.x;
  float thr = th[r], s = 0.0f;
  for (int c = lane; c < NTOT; c += 32) {
    float w = W[(size_t)r * NTOT + c];
    float m = fminf(thr, th[c]);        // keep if in top-k of row i OR row j
    w = (w >= m) ? w : 0.0f;
    W[(size_t)r * NTOT + c] = w;
    s += w;
  }
  s = wave_sum(s);
  if (lane == 0) deg[r] = s;
}

__global__ void k_dsi(const float* __restrict__ deg, float* __restrict__ dsi) {
  int i = blockIdx.x * blockDim.x + threadIdx.x;
  if (i < NTOT) dsi[i] = rsqrtf(deg[i] + EPSF);
}

__global__ void k_make_s16(const float* __restrict__ W, const float* __restrict__ dsi,
                           _Float16* __restrict__ S16) {
  size_t i = (size_t)blockIdx.x * blockDim.x + threadIdx.x;
  int r = (int)(i >> 12), c = (int)(i & 4095);
  S16[i] = (_Float16)(dsi[r] * W[i] * dsi[c]);
}

// ---------------- Stage 5: Chebyshev solve ----------------------------------
__global__ void k_init_xy(const float* __restrict__ slab, float* __restrict__ Y,
                          float* __restrict__ X, float* __restrict__ dX) {
  int i = blockIdx.x * blockDim.x + threadIdx.x;   // NTOT*NCLS
  int r = i >> 6;
  float y = (r < NSUPR) ? slab[i] : 0.0f;
  Y[i] = y; X[i] = y; dX[i] = y;   // x1 = dx0 = r0/theta = y (theta = 1)
}

__global__ void k_xt(const float* __restrict__ X, _Float16* __restrict__ Xt) {
  int i = blockIdx.x * blockDim.x + threadIdx.x;   // NTOT*NCLS
  int r = i >> 6, c = i & 63;
  Xt[(size_t)c * NTOT + r] = (_Float16)X[i];
}

// T = S @ X : per-wave 32x32 tile; block = 4 waves covers 64 rows x 64 cols.
__global__ __launch_bounds__(128)
void k_spmm(const _Float16* __restrict__ S, const _Float16* __restrict__ Xt,
            float* __restrict__ T) {
  int lane = threadIdx.x & 31, wave = threadIdx.x >> 5;
  int m0 = blockIdx.x * 64 + (wave >> 1) * 32;
  int n0 = (wave & 1) * 32;
  v8f acc[2][2] = {};
  for (int k0 = 0; k0 < NTOT; k0 += 32) {
    v16h a0 = load_frag(S,  NTOT, m0,      k0, lane);
    v16h a1 = load_frag(S,  NTOT, m0 + 16, k0, lane);
    v16h b0 = load_frag(Xt, NTOT, n0,      k0, lane);
    v16h b1 = load_frag(Xt, NTOT, n0 + 16, k0, lane);
    WMMA_F16(acc[0][0], a0, b0);
    WMMA_F16(acc[0][1], a0, b1);
    WMMA_F16(acc[1][0], a1, b0);
    WMMA_F16(acc[1][1], a1, b1);
  }
  for (int tm = 0; tm < 2; ++tm) {
    int rowb = m0 + tm * 16 + ((lane >> 4) << 3);
    for (int tn = 0; tn < 2; ++tn) {
      int col = n0 + tn * 16 + (lane & 15);
      for (int r = 0; r < 8; ++r)
        T[(size_t)(rowb + r) * NCLS + col] = acc[tm][tn][r];
    }
  }
}

// r = y - (X - alpha*T);  dX = c1*dX + c2*r;  X += dX
__global__ void k_update(float* __restrict__ X, float* __restrict__ dX,
                         const float* __restrict__ T, const float* __restrict__ Y,
                         float c1, float c2) {
  int i = blockIdx.x * blockDim.x + threadIdx.x;
  float r = Y[i] - X[i] + ALPHA_ * T[i];
  float d = c1 * dX[i] + c2 * r;
  dX[i] = d;
  X[i] += d;
}

// ---------------- Stage 6: loss / acc / outputs -----------------------------
__global__ __launch_bounds__(32)
void k_loss(const float* __restrict__ X, const float* __restrict__ slab,
            const float* __restrict__ qlab, float* __restrict__ rowloss,
            float* __restrict__ qc) {
  int r = blockIdx.x, lane = threadIdx.x;
  const float* lab = (r < NSUPR) ? (slab + (size_t)r * NCLS)
                                 : (qlab + (size_t)(r - NSUPR) * NCLS);
  float lv = -1.0f; int li = NCLS;
  for (int c = lane; c < NCLS; c += 32) {
    float x = lab[c];
    if (x > lv) { lv = x; li = c; }
  }
  wave_argmax(lv, li);
  int gt = li;

  const float* f = X + (size_t)r * NCLS;
  float mv = -3.4e38f; int mi = NCLS;
  for (int c = lane; c < NCLS; c += 32) {
    float x = f[c];
    if (x > mv) { mv = x; mi = c; }
  }
  wave_argmax(mv, mi);
  float se = 0.0f;
  for (int c = lane; c < NCLS; c += 32) se += __expf(f[c] - mv);
  se = wave_sum(se);
  if (lane == 0) {
    rowloss[r] = -(f[gt] - mv - __logf(se));
    if (r >= NSUPR) qc[r - NSUPR] = (mi == gt) ? 1.0f : 0.0f;
  }
}

__global__ __launch_bounds__(256)
void k_finalize(const float* __restrict__ rowloss, const float* __restrict__ qc,
                float* __restrict__ out) {
  __shared__ float sm[256];
  int t = threadIdx.x;
  float s = 0.0f;
  for (int i = t; i < NTOT; i += 256) s += rowloss[i];
  sm[t] = s; __syncthreads();
  for (int o = 128; o > 0; o >>= 1) { if (t < o) sm[t] += sm[t + o]; __syncthreads(); }
  if (t == 0) out[0] = sm[0] / (float)NTOT;
  __syncthreads();
  float a = 0.0f;
  for (int i = t; i < NQRYR; i += 256) a += qc[i];
  sm[t] = a; __syncthreads();
  for (int o = 128; o > 0; o >>= 1) { if (t < o) sm[t] += sm[t + o]; __syncthreads(); }
  if (t == 0) out[1] = sm[0] / (float)NQRYR;
}

__global__ void k_copy_fq(const float* __restrict__ X, float* __restrict__ out) {
  int i = blockIdx.x * blockDim.x + threadIdx.x;   // NQRYR*NCLS
  out[2 + i] = X[(size_t)NSUPR * NCLS + i];
}

// ---------------------------------------------------------------------------
static inline size_t al256(size_t x) { return (x + 255) & ~(size_t)255; }

extern "C" void kernel_launch(void* const* d_in, const int* in_sizes, int n_in,
                              void* d_out, int out_size, void* d_ws, size_t ws_size,
                              hipStream_t stream) {
  const float* emb   = (const float*)d_in[0];   // 4096 x 1600
  const float* sigma = (const float*)d_in[1];   // 4096 x 1
  const float* slab  = (const float*)d_in[2];   // 1024 x 64
  const float* qlab  = (const float*)d_in[3];   // 3072 x 64
  float* out = (float*)d_out;                   // [loss, acc, Fq(3072x64)]

  // workspace layout (~118 MB total)
  char* ws = (char*)d_ws;
  size_t o = 0;
  _Float16* E16 = (_Float16*)(ws + o); o = al256(o + (size_t)NTOT * DFEAT * 2);
  float*    sq  = (float*)   (ws + o); o = al256(o + (size_t)NTOT * 4);
  float*    W   = (float*)   (ws + o); o = al256(o + (size_t)NTOT * NTOT * 4);
  float*    th  = (float*)   (ws + o); o = al256(o + (size_t)NTOT * 4);
  float*    deg = (float*)   (ws + o); o = al256(o + (size_t)NTOT * 4);
  float*    dsi = (float*)   (ws + o); o = al256(o + (size_t)NTOT * 4);
  _Float16* S16 = (_Float16*)(ws + o); o = al256(o + (size_t)NTOT * NTOT * 2);
  float*    Y   = (float*)   (ws + o); o = al256(o + (size_t)NTOT * NCLS * 4);
  float*    X   = (float*)   (ws + o); o = al256(o + (size_t)NTOT * NCLS * 4);
  float*    dX  = (float*)   (ws + o); o = al256(o + (size_t)NTOT * NCLS * 4);
  float*    T   = (float*)   (ws + o); o = al256(o + (size_t)NTOT * NCLS * 4);
  _Float16* Xt  = (_Float16*)(ws + o); o = al256(o + (size_t)NTOT * NCLS * 2);
  float*    rls = (float*)   (ws + o); o = al256(o + (size_t)NTOT * 4);
  float*    qc  = (float*)   (ws + o); o = al256(o + (size_t)NQRYR * 4);
  (void)ws_size; (void)in_sizes; (void)n_in; (void)out_size;

  // 1) normalize + f16 + row squared norms
  k_norm<<<NTOT, 32, 0, stream>>>(emb, sigma, E16, sq);
  // 2) Gram GEMM (WMMA) fused into W = exp(-dist/2)
  k_gram_w<<<dim3(NTOT / 64, NTOT / 64), 128, 0, stream>>>(E16, sq, W);
  // 3) exact top-20 threshold per row
  k_topk<<<NTOT, 32, 0, stream>>>(W, th);
  // 4) symmetric mask + degrees -> S in f16
  k_mask_deg<<<NTOT, 32, 0, stream>>>(W, th, deg);
  k_dsi<<<NTOT / 256, 256, 0, stream>>>(deg, dsi);
  k_make_s16<<<(unsigned)(((size_t)NTOT * NTOT) / 256), 256, 0, stream>>>(W, dsi, S16);
  // 5) Chebyshev semi-iteration for (I - alpha S) X = Y
  k_init_xy<<<(NTOT * NCLS) / 256, 256, 0, stream>>>(slab, Y, X, dX);
  {
    const double theta = 1.0, delta = (double)ALPHA_;   // spectrum [1-a, 1+a]
    const double sigma1 = theta / delta;
    double rho_prev = 1.0 / sigma1;
    for (int k = 1; k <= CHEB_ITERS; ++k) {
      double rho = 1.0 / (2.0 * sigma1 - rho_prev);
      float c1 = (float)(rho * rho_prev);
      float c2 = (float)(2.0 * rho / delta);
      k_xt<<<(NTOT * NCLS) / 256, 256, 0, stream>>>(X, Xt);
      k_spmm<<<NTOT / 64, 128, 0, stream>>>(S16, Xt, T);
      k_update<<<(NTOT * NCLS) / 256, 256, 0, stream>>>(X, dX, T, Y, c1, c2);
      rho_prev = rho;
    }
  }
  // 6) loss / acc / Fq
  k_loss<<<NTOT, 32, 0, stream>>>(X, slab, qlab, rls, qc);
  k_finalize<<<1, 256, 0, stream>>>(rls, qc, out);
  k_copy_fq<<<(NQRYR * NCLS) / 256, 256, 0, stream>>>(X, out);
}
